// MultiQueryAttention_33827162423862
// MI455X (gfx1250) — matrix operand order repaired and checked by reading, not compiled, not run on hardware
//
#include <hip/hip_runtime.h>
#include <hip/hip_bf16.h>

#define B_  2
#define S_  2048
#define D_  2048
#define H_  16
#define HD_ 128

typedef _Float16 v16h __attribute__((ext_vector_type(16)));
typedef float    v8f  __attribute__((ext_vector_type(8)));

union Frag16 { v16h v; uint4 q[2]; };
union H8     { uint4 q; _Float16 h[8]; };
union H4     { uint2 u; _Float16 h[4]; };

__device__ __forceinline__ v8f vzero8() {
    v8f z = {0.f,0.f,0.f,0.f,0.f,0.f,0.f,0.f};
    return z;
}

__device__ __forceinline__ v8f wmma16(v16h a, v16h b, v8f c) {
    // D = A(16x32 f16) * B(32x16 f16) + C(16x16 f32)
    return __builtin_amdgcn_wmma_f32_16x16x32_f16(
        /*neg_a=*/false, a, /*neg_b=*/false, b,
        /*c_mod=*/(short)0, c, /*reuse_a=*/false, /*reuse_b=*/false);
}

// ---------------------------------------------------------------------------
// CDNA5 async global->LDS copy (ASYNCcnt-tracked). LDS offset = low 32 bits of
// the generic shared pointer (shared aperture lives in addr[63:32]).
// ---------------------------------------------------------------------------
__device__ __forceinline__ void async_copy_b128(const void* gsrc, void* ldst) {
    unsigned lofs = (unsigned)(uintptr_t)ldst;
    unsigned long long g = (unsigned long long)(uintptr_t)gsrc;
    asm volatile("global_load_async_to_lds_b128 %0, %1, off"
                 :: "v"(lofs), "v"(g) : "memory");
}
__device__ __forceinline__ void wait_async0() {
#if __has_builtin(__builtin_amdgcn_s_wait_asynccnt)
    __builtin_amdgcn_s_wait_asynccnt(0);
#else
    asm volatile("s_wait_asynccnt 0x0" ::: "memory");
#endif
}

// ---------------------------------------------------------------------------
// 16-lane (DPP-row) butterfly reductions: xor1, xor2, half-mirror, mirror.
// Pure VALU; never crosses the 16-lane row that holds one D-fragment row.
// ---------------------------------------------------------------------------
template <int CTRL>
__device__ __forceinline__ float dpp_f32(float x) {
    int xi = __float_as_int(x);
    int r = __builtin_amdgcn_update_dpp(xi, xi, CTRL, 0xf, 0xf, true);
    return __int_as_float(r);
}
__device__ __forceinline__ float rowmax16(float x) {
    x = fmaxf(x, dpp_f32<0xB1>(x));   // quad_perm [1,0,3,2]
    x = fmaxf(x, dpp_f32<0x4E>(x));   // quad_perm [2,3,0,1]
    x = fmaxf(x, dpp_f32<0x141>(x));  // row_half_mirror
    x = fmaxf(x, dpp_f32<0x140>(x));  // row_mirror
    return x;
}
__device__ __forceinline__ float rowsum16(float x) {
    x += dpp_f32<0xB1>(x);
    x += dpp_f32<0x4E>(x);
    x += dpp_f32<0x141>(x);
    x += dpp_f32<0x140>(x);
    return x;
}

// ---------------------------------------------------------------------------
// fp32 -> fp16 conversion, 4 elements/thread
// ---------------------------------------------------------------------------
__global__ void cvt4_f32_f16(const float4* __restrict__ in,
                             _Float16* __restrict__ out, int n4) {
    int i = blockIdx.x * blockDim.x + threadIdx.x;
    if (i < n4) {
        float4 v = in[i];
        H4 o;
        o.h[0] = (_Float16)v.x; o.h[1] = (_Float16)v.y;
        o.h[2] = (_Float16)v.z; o.h[3] = (_Float16)v.w;
        *(uint2*)&out[i * 4] = o.u;
    }
}

// ---------------------------------------------------------------------------
// Tiled WMMA GEMM: C[M,N] = A[M,K] @ B[K,N] + bias[N]
// Block tile 128(M) x 128(N), K-step 32, 256 threads = 8 waves.
// Wave w owns rows w*16..w*16+15 and all 8 n-subtiles (8 accumulators).
// A tile staged with async global->LDS; B tile staged transposed manually.
// ---------------------------------------------------------------------------
template <typename OutT>
__global__ __launch_bounds__(256)
void gemm_wmma_f16(const _Float16* __restrict__ A,
                   const _Float16* __restrict__ Bm,
                   const float* __restrict__ bias,
                   OutT* __restrict__ C, int M, int N, int K) {
    __shared__ __align__(16) _Float16 ldsA[128 * 32];   // row-major, stride 32
    __shared__ __align__(16) _Float16 ldsBt[128 * 32];  // transposed: [n][k]

    const int t    = threadIdx.x;
    const int w    = t >> 5;
    const int lane = t & 31;
    const int lrow = lane & 15;
    const int hi   = lane >> 4;
    const int m0   = blockIdx.y * 128;
    const int n0   = blockIdx.x * 128;

    v8f acc[8];
#pragma unroll
    for (int nt = 0; nt < 8; ++nt) acc[nt] = vzero8();

    for (int k0 = 0; k0 < K; k0 += 32) {
        __syncthreads();  // previous iteration done reading ldsA/ldsBt
        // --- stage A tile: 128x32 halves via async copies (2 chunks/thread) ---
#pragma unroll
        for (int i = 0; i < 2; ++i) {
            int c = t + i * 256;
            int arow = c >> 2, k8 = (c & 3) << 3;
            async_copy_b128(&A[(size_t)(m0 + arow) * K + k0 + k8],
                            &ldsA[arow * 32 + k8]);
        }
        // --- stage B tile transposed: 32x128 halves (2 chunks/thread) ---
#pragma unroll
        for (int i = 0; i < 2; ++i) {
            int c = t + i * 256;
            int krow = c >> 4, n8 = (c & 15) << 3;
            const _Float16* src = &Bm[(size_t)(k0 + krow) * N + n0 + n8];
            H8 u; u.q = *(const uint4*)src;
            if (k0 + 32 < K) __builtin_prefetch(src + (size_t)32 * N, 0, 1);
#pragma unroll
            for (int j = 0; j < 8; ++j) ldsBt[(n8 + j) * 32 + krow] = u.h[j];
        }
        wait_async0();
        __syncthreads();

        // A fragment: row = w*16+lrow, K chunks {hi*8..+7, 16+hi*8..+7}
        Frag16 af;
        af.q[0] = *(const uint4*)&ldsA[(w * 16 + lrow) * 32 + hi * 8];
        af.q[1] = *(const uint4*)&ldsA[(w * 16 + lrow) * 32 + 16 + hi * 8];
#pragma unroll
        for (int nt = 0; nt < 8; ++nt) {
            // B fragment: col = nt*16+lrow, K chunks {hi*8..+7, 16+hi*8..+7}
            int nl = nt * 16 + lrow;
            Frag16 bf;
            bf.q[0] = *(const uint4*)&ldsBt[nl * 32 + hi * 8];
            bf.q[1] = *(const uint4*)&ldsBt[nl * 32 + 16 + hi * 8];
            acc[nt] = wmma16(af.v, bf.v, acc[nt]);
        }
    }

    // epilogue: D layout row = i + hi*8, col = lane&15
#pragma unroll
    for (int nt = 0; nt < 8; ++nt) {
        int col = n0 + nt * 16 + lrow;
        float bv = bias[col];
#pragma unroll
        for (int i = 0; i < 8; ++i) {
            int row = m0 + w * 16 + i + hi * 8;
            C[(size_t)row * N + col] = (OutT)(acc[nt][i] + bv);
        }
    }
}

// ---------------------------------------------------------------------------
// Flash attention: one block = 128 query rows of one (b,h).
// 8 waves, wave w owns query rows w*16..w*16+15 (softmax stays intra-wave,
// reductions via DPP row butterflies).
// K-block = 32 keys/iteration: S=QK^T (8 WMMA), online softmax, O+=PV (8 WMMA).
// ---------------------------------------------------------------------------
__global__ __launch_bounds__(256)
void flash_attn_wmma(const _Float16* __restrict__ Q,
                     const _Float16* __restrict__ Kh,
                     const _Float16* __restrict__ Vh,
                     const float* __restrict__ mask,
                     _Float16* __restrict__ O) {
    __shared__ __align__(16) _Float16 ldsK[32 * 128];   // [krow][d] == B^T layout
    __shared__ __align__(16) _Float16 ldsVt[128 * 32];  // [d][krow]
    __shared__ __align__(16) _Float16 ldsP[128 * 32];   // [qrow_local][krow]

    const int t    = threadIdx.x;
    const int w    = t >> 5;
    const int lane = t & 31;
    const int lrow = lane & 15;
    const int hi   = lane >> 4;

    int bid = blockIdx.x;
    const int qb = bid & 15; bid >>= 4;
    const int h  = bid & 15; bid >>= 4;
    const int b  = bid;

    const int qrow0 = qb * 128 + w * 16;

    // Q fragments for this wave's 16 rows, all 128 d, held in registers.
    Frag16 qf[4];
    const _Float16* qbase = Q + ((size_t)(b * S_ + qrow0 + lrow)) * D_ + h * HD_;
#pragma unroll
    for (int ks = 0; ks < 4; ++ks) {
        qf[ks].q[0] = *(const uint4*)&qbase[ks * 32 + hi * 8];
        qf[ks].q[1] = *(const uint4*)&qbase[ks * 32 + 16 + hi * 8];
    }

    v8f Of[8];
#pragma unroll
    for (int o = 0; o < 8; ++o) Of[o] = vzero8();
    float mrow[8], lsum[8];
#pragma unroll
    for (int i = 0; i < 8; ++i) { mrow[i] = -1e30f; lsum[i] = 0.f; }

    const float scale = 0.08838834764831845f;  // 1/sqrt(128)

    for (int kb = 0; kb < S_; kb += 32) {
        __syncthreads();  // previous iteration done with ldsK/ldsVt
        // stage K (row-major, async) and V^T (manual transpose)
#pragma unroll
        for (int i = 0; i < 2; ++i) {
            int c = t + i * 256;
            int krow = c >> 4, d8 = (c & 15) << 3;
            size_t grow = (size_t)(b * S_ + kb + krow) * HD_ + d8;
            async_copy_b128(&Kh[grow], &ldsK[krow * 128 + d8]);
            H8 u; u.q = *(const uint4*)&Vh[grow];
#pragma unroll
            for (int j = 0; j < 8; ++j) ldsVt[(d8 + j) * 32 + krow] = u.h[j];
        }
        wait_async0();
        __syncthreads();

        // S tile 16x32 for this wave: two 16x16 WMMA accumulators
        v8f sacc[2];
        sacc[0] = vzero8(); sacc[1] = vzero8();
#pragma unroll
        for (int nt = 0; nt < 2; ++nt) {
            int nl = nt * 16 + lrow;
#pragma unroll
            for (int ks = 0; ks < 4; ++ks) {
                Frag16 bf;  // column of K^T = row of K: contiguous in ldsK
                bf.q[0] = *(const uint4*)&ldsK[nl * 128 + ks * 32 + hi * 8];
                bf.q[1] = *(const uint4*)&ldsK[nl * 128 + ks * 32 + 16 + hi * 8];
                sacc[nt] = wmma16(qf[ks].v, bf.v, sacc[nt]);
            }
        }

        // scale + additive mask
        float s0[8], s1[8];
#pragma unroll
        for (int i = 0; i < 8; ++i) {
            int qr = qb * 128 + w * 16 + i + hi * 8;
            const float* mp = mask + ((size_t)b * S_ + qr) * S_ + kb;
            s0[i] = sacc[0][i] * scale + mp[lrow];
            s1[i] = sacc[1][i] * scale + mp[16 + lrow];
        }

        // online softmax (row lives in one 16-lane DPP row)
        float alpha[8], p0[8], p1[8];
#pragma unroll
        for (int i = 0; i < 8; ++i) {
            float mx = rowmax16(fmaxf(s0[i], s1[i]));
            float mnew = fmaxf(mrow[i], mx);
            alpha[i] = __expf(mrow[i] - mnew);
            p0[i] = __expf(s0[i] - mnew);
            p1[i] = __expf(s1[i] - mnew);
            float rs = rowsum16(p0[i] + p1[i]);
            lsum[i] = lsum[i] * alpha[i] + rs;
            mrow[i] = mnew;
        }

        // rescale running O
#pragma unroll
        for (int o = 0; o < 8; ++o)
#pragma unroll
            for (int i = 0; i < 8; ++i) Of[o][i] = Of[o][i] * alpha[i];

        // P: D-layout -> LDS row-major (wave-private rows; DS in-order per wave)
#pragma unroll
        for (int i = 0; i < 8; ++i) {
            int pr = w * 16 + i + hi * 8;
            ldsP[pr * 32 + lrow]      = (_Float16)p0[i];
            ldsP[pr * 32 + 16 + lrow] = (_Float16)p1[i];
        }
        // reload P in A-fragment layout
        Frag16 pf;
        pf.q[0] = *(const uint4*)&ldsP[(w * 16 + lrow) * 32 + hi * 8];
        pf.q[1] = *(const uint4*)&ldsP[(w * 16 + lrow) * 32 + 16 + hi * 8];

        // O += P @ V  (V^T staged: B fragment contiguous)
#pragma unroll
        for (int o = 0; o < 8; ++o) {
            int nl = o * 16 + lrow;
            Frag16 bf;
            bf.q[0] = *(const uint4*)&ldsVt[nl * 32 + hi * 8];
            bf.q[1] = *(const uint4*)&ldsVt[nl * 32 + 16 + hi * 8];
            Of[o] = wmma16(pf.v, bf.v, Of[o]);
        }
    }

    // normalize and store into concat-head layout [B*S, D]
#pragma unroll
    for (int i = 0; i < 8; ++i) {
        float inv = 1.f / lsum[i];
        int qr = qb * 128 + w * 16 + i + hi * 8;
        _Float16* orow = O + ((size_t)(b * S_ + qr)) * D_ + h * HD_;
#pragma unroll
        for (int o = 0; o < 8; ++o)
            orow[o * 16 + lrow] = (_Float16)(Of[o][i] * inv);
    }
}

// ---------------------------------------------------------------------------
extern "C" void kernel_launch(void* const* d_in, const int* in_sizes, int n_in,
                              void* d_out, int out_size, void* d_ws, size_t ws_size,
                              hipStream_t stream) {
    const float* x    = (const float*)d_in[0];
    const float* mask = (const float*)d_in[1];
    const float* wq_w = (const float*)d_in[2];
    const float* wq_b = (const float*)d_in[3];
    const float* wk_w = (const float*)d_in[4];
    const float* wk_b = (const float*)d_in[5];
    const float* wv_w = (const float*)d_in[6];
    const float* wv_b = (const float*)d_in[7];
    const float* wo_w = (const float*)d_in[8];
    const float* wo_b = (const float*)d_in[9];
    float* out = (float*)d_out;

    char* ws = (char*)d_ws;
    size_t off = 0;
    auto alloc = [&](size_t bytes) {
        void* p = ws + off;
        off = (off + bytes + 255) & ~(size_t)255;
        return p;
    };
    const size_t nX  = (size_t)B_ * S_ * D_;   // 8,388,608
    const size_t nWq = (size_t)D_ * D_;        // 4,194,304
    const size_t nWk = (size_t)D_ * HD_;       // 262,144
    const size_t nKV = (size_t)B_ * S_ * HD_;  // 524,288

    _Float16* xh  = (_Float16*)alloc(nX * 2);
    _Float16* wqh = (_Float16*)alloc(nWq * 2);
    _Float16* wkh = (_Float16*)alloc(nWk * 2);
    _Float16* wvh = (_Float16*)alloc(nWk * 2);
    _Float16* woh = (_Float16*)alloc(nWq * 2);
    _Float16* Qh  = (_Float16*)alloc(nX * 2);
    _Float16* Kf  = (_Float16*)alloc(nKV * 2);
    _Float16* Vf  = (_Float16*)alloc(nKV * 2);
    _Float16* Ah  = (_Float16*)alloc(nX * 2);

    auto cvt = [&](const float* src, _Float16* dst, size_t n) {
        int n4 = (int)(n / 4);
        cvt4_f32_f16<<<dim3(n4 / 256), dim3(256), 0, stream>>>((const float4*)src, dst, n4);
    };
    cvt(x, xh, nX);
    cvt(wq_w, wqh, nWq);
    cvt(wk_w, wkh, nWk);
    cvt(wv_w, wvh, nWk);
    cvt(wo_w, woh, nWq);

    const int M = B_ * S_;  // 4096
    // Q/K/V projections
    gemm_wmma_f16<_Float16><<<dim3(D_ / 128, M / 128), dim3(256), 0, stream>>>(
        xh, wqh, wq_b, Qh, M, D_, D_);
    gemm_wmma_f16<_Float16><<<dim3(HD_ / 128, M / 128), dim3(256), 0, stream>>>(
        xh, wkh, wk_b, Kf, M, HD_, D_);
    gemm_wmma_f16<_Float16><<<dim3(HD_ / 128, M / 128), dim3(256), 0, stream>>>(
        xh, wvh, wv_b, Vf, M, HD_, D_);

    // attention: grid = B*H*(S/128)
    flash_attn_wmma<<<dim3(B_ * H_ * (S_ / 128)), dim3(256), 0, stream>>>(
        Qh, Kf, Vf, mask, Ah);

    // output projection (fp32 out)
    gemm_wmma_f16<float><<<dim3(D_ / 128, M / 128), dim3(256), 0, stream>>>(
        Ah, woh, wo_b, out, M, D_, D_);
}
